// nBEFRC_40278203302325
// MI455X (gfx1250) — compile-verified, matrix operand order checked by reading, not run on hardware
//
#include <hip/hip_runtime.h>
#include <stdint.h>

#define B_      64
#define L_      1024
#define IN_     128
#define MEM_    512
#define NWG     32
#define SLICE   16
#define THREADS 128
#define NW      4
#define DT_     0.1f

typedef __attribute__((ext_vector_type(8)))  float  v8f;
typedef __attribute__((ext_vector_type(8)))  __bf16 v8bf;
typedef __attribute__((ext_vector_type(16))) __bf16 v16bf;

__device__ __forceinline__ unsigned short f2bf(float f) {
  unsigned u = __float_as_uint(f);
  u += 0x7fffu + ((u >> 16) & 1u);
  return (unsigned short)(u >> 16);
}

__device__ __forceinline__ float sigm(float x) {
  return 1.0f / (1.0f + __expf(-x));
}
__device__ __forceinline__ float ftanh(float x) {
  float ax = fabsf(x);
  float e  = __expf(-2.0f * ax);
  float t  = (1.0f - e) / (1.0f + e);
  return copysignf(t, x);
}

__device__ __forceinline__ void grid_sync(unsigned* cnt, unsigned phase) {
  __syncthreads();
  if (threadIdx.x == 0) {
    __hip_atomic_fetch_add(cnt, 1u, __ATOMIC_RELEASE, __HIP_MEMORY_SCOPE_AGENT);
    while (__hip_atomic_load(cnt, __ATOMIC_ACQUIRE, __HIP_MEMORY_SCOPE_AGENT) <
           (unsigned)NWG * phase) {
      __builtin_amdgcn_s_sleep(8);
    }
  }
  __syncthreads();
}

__global__ __launch_bounds__(THREADS, 1)
void fastslow_rnn_kernel(const float* __restrict__ u,
                         const float* __restrict__ h0,
                         const float* __restrict__ Wi,
                         const float* __restrict__ bi,
                         const float* __restrict__ Wm,
                         const float* __restrict__ bm,
                         float* __restrict__ out,
                         unsigned short* __restrict__ stage,  // 2 ping-pong bf16 [64*512]
                         unsigned* __restrict__ cnt) {
  // LDS: 64 + 16 + 80 + 24 = 184 KB (CDNA5 WGP has 320 KB)
  __shared__ __attribute__((aligned(16))) unsigned short lds_hf[B_ * MEM_];        // bf16 hf_t
  __shared__ __attribute__((aligned(16))) unsigned short lds_u [B_ * IN_];         // bf16 u_t
  __shared__ __attribute__((aligned(32))) unsigned short lds_wm[16 * 5 * 32 * 16]; // Wm frags
  __shared__ __attribute__((aligned(32))) unsigned short lds_wi[ 4 * 6 * 32 * 16]; // Wi frags

  const int tid   = threadIdx.x;
  const int wave  = tid >> 5;
  const int lane  = tid & 31;
  const int wg    = blockIdx.x;
  const int j0    = wg * SLICE;
  const int col   = lane & 15;
  const int khalf = lane >> 4;
  const int mrow  = wave * 16 + col;  // A-fragment row handled by this lane

  // ---- one-time: swizzle Wm/Wi slices into LDS in exact B-fragment layout (bf16) ----
  for (int blk = wave; blk < 16 * 5; blk += NW) {     // blk = kt*5 + g
    int kt = blk / 5, g = blk % 5;
    int n  = g * MEM_ + j0 + col;
    unsigned short* dst = &lds_wm[(blk * 32 + lane) * 16];
#pragma unroll
    for (int h = 0; h < 16; ++h) {
      int K = kt * 32 + khalf * 8 + (h & 7) + ((h >> 3) * 16);
      dst[h] = f2bf(Wm[(size_t)K * (5 * MEM_) + n]);
    }
  }
  for (int blk = wave; blk < 4 * 6; blk += NW) {      // blk = kt*6 + g
    int kt = blk / 6, g = blk % 6;
    int n  = g * MEM_ + j0 + col;
    unsigned short* dst = &lds_wi[(blk * 32 + lane) * 16];
#pragma unroll
    for (int h = 0; h < 16; ++h) {
      int K = kt * 32 + khalf * 8 + (h & 7) + ((h >> 3) * 16);
      dst[h] = f2bf(Wi[(size_t)K * (6 * MEM_) + n]);
    }
  }

  // per-lane biases (j = j0+col fixed across accumulator regs)
  float biL[6], bmL[5];
#pragma unroll
  for (int g = 0; g < 6; ++g) biL[g] = bi[g * MEM_ + j0 + col];
#pragma unroll
  for (int g = 0; g < 5; ++g) bmL[g] = bm[g * MEM_ + j0 + col];

  // persistent state in WMMA C/D layout: reg r -> b = wave*16 + r + 8*khalf, j = j0+col
  float hf_t[8], hs_t[8];
#pragma unroll
  for (int r = 0; r < 8; ++r) {
    int b   = wave * 16 + r + 8 * khalf;
    hf_t[r] = h0[(size_t)b * MEM_ + j0 + col];
    hs_t[r] = h0[(size_t)(B_ + b) * MEM_ + j0 + col];
    stage[(size_t)b * MEM_ + j0 + col] = f2bf(hf_t[r]);  // seed ping buffer 0
  }

  grid_sync(cnt, 1);

  for (int t = 0; t < L_; ++t) {
    // ---- async copy full hf (bf16, 64 KB) from L2 staging into LDS ----
    const unsigned short* src = stage + (size_t)(t & 1) * (B_ * MEM_);
    for (int i = tid; i < (B_ * MEM_) / 8; i += THREADS) {
      unsigned ldsoff          = (unsigned)(uintptr_t)&lds_hf[i * 8];
      const unsigned short* gp = src + i * 8;
      asm volatile("global_load_async_to_lds_b128 %0, %1, off"
                   :: "v"(ldsoff), "v"(gp) : "memory");
    }
    // ---- stage u_t as bf16 (fp32 global -> bf16 LDS), coalesced float4 ----
    for (int i4 = tid * 4; i4 < B_ * IN_; i4 += THREADS * 4) {
      int b = i4 >> 7, k = i4 & 127;
      const float4 f = *(const float4*)(u + ((size_t)b * L_ + t) * IN_ + k);
      ushort4 s;
      s.x = f2bf(f.x); s.y = f2bf(f.y); s.z = f2bf(f.z); s.w = f2bf(f.w);
      *(ushort4*)&lds_u[i4] = s;
    }
    asm volatile("s_wait_asynccnt 0" ::: "memory");
    __syncthreads();

    v8f accM[5], accI[6];
    const v8f z = {0.f, 0.f, 0.f, 0.f, 0.f, 0.f, 0.f, 0.f};
#pragma unroll
    for (int g = 0; g < 5; ++g) accM[g] = z;
#pragma unroll
    for (int g = 0; g < 6; ++g) accI[g] = z;

    const __bf16* hfb = (const __bf16*)lds_hf;
    const __bf16* ub  = (const __bf16*)lds_u;

    // me = hf @ Wm : K = 512 -> 16 k-tiles x 5 gates
    for (int kt = 0; kt < MEM_ / 32; ++kt) {
      const int kb = kt * 32 + khalf * 8;
      v8bf lo = *(const v8bf*)&hfb[mrow * MEM_ + kb];
      v8bf hi = *(const v8bf*)&hfb[mrow * MEM_ + kb + 16];
      v16bf a = __builtin_shufflevector(lo, hi, 0, 1, 2, 3, 4, 5, 6, 7,
                                                8, 9, 10, 11, 12, 13, 14, 15);
#pragma unroll
      for (int g = 0; g < 5; ++g) {
        v16bf bb = *(const v16bf*)&lds_wm[((kt * 5 + g) * 32 + lane) * 16];
        accM[g] = __builtin_amdgcn_wmma_f32_16x16x32_bf16(
            false, a, false, bb, (short)0, accM[g], false, false);
      }
    }
    // ie_t = u_t @ Wi : K = 128 -> 4 k-tiles x 6 gates
    for (int kt = 0; kt < IN_ / 32; ++kt) {
      const int kb = kt * 32 + khalf * 8;
      v8bf lo = *(const v8bf*)&ub[mrow * IN_ + kb];
      v8bf hi = *(const v8bf*)&ub[mrow * IN_ + kb + 16];
      v16bf a = __builtin_shufflevector(lo, hi, 0, 1, 2, 3, 4, 5, 6, 7,
                                                8, 9, 10, 11, 12, 13, 14, 15);
#pragma unroll
      for (int g = 0; g < 6; ++g) {
        v16bf bb = *(const v16bf*)&lds_wi[((kt * 6 + g) * 32 + lane) * 16];
        accI[g] = __builtin_amdgcn_wmma_f32_16x16x32_bf16(
            false, a, false, bb, (short)0, accI[g], false, false);
      }
    }

    // ---- elementwise gates directly in the accumulator register layout ----
    unsigned short* dstg = stage + (size_t)((t + 1) & 1) * (B_ * MEM_);
#pragma unroll
    for (int r = 0; r < 8; ++r) {
      float ia  = accI[0][r] + biL[0];
      float ibv = accI[1][r] + biL[1];
      float ic  = accI[2][r] + biL[2];
      float id  = accI[3][r] + biL[3];
      float ieg = accI[4][r] + biL[4];
      float io  = accI[5][r] + biL[5];
      float ma  = accM[0][r] + bmL[0];
      float mb  = accM[1][r] + bmL[1];
      float mc  = accM[2][r] + bmL[2];
      float md  = accM[3][r] + bmL[3];
      float me_ = accM[4][r] + bmL[4];

      float ga = 1.0f + ftanh(ia + ma);
      float gb = 1.5f + 1.5f * ftanh(ibv + mb);
      float gc = 3.0f * DT_ + (1.0f - 3.0f * DT_) * sigm(ic + mc);
      float gd = 0.3f * DT_ * sigm(id + md);
      float ge = 1.0f + sigm(ieg + me_);

      float hf = hf_t[r], hs = hs_t[r];
      float hfn = (1.0f - gc) * hf + gc * ftanh(io + (ga + gb * hf * hf - hs) * hf);
      float eh  = ge * hf;
      float eh2 = eh * eh;
      float hsn = hs * (1.0f - gd) + gd * eh2 * eh2;
      hf_t[r] = hfn;
      hs_t[r] = hsn;

      int b = wave * 16 + r + 8 * khalf;
      int j = j0 + col;
      out[((size_t)b * L_ + t) * MEM_ + j] = hfn;          // h_t[b][t][j]
      dstg[(size_t)b * MEM_ + j] = f2bf(hfn);              // next-step staging
      if (t == L_ - 1)
        out[(size_t)B_ * L_ * MEM_ + (size_t)b * MEM_ + j] = hfn;  // hf_last
    }

    grid_sync(cnt, (unsigned)(t + 2));
  }
}

extern "C" void kernel_launch(void* const* d_in, const int* in_sizes, int n_in,
                              void* d_out, int out_size, void* d_ws, size_t ws_size,
                              hipStream_t stream) {
  const float* u  = (const float*)d_in[0];
  const float* h0 = (const float*)d_in[1];
  const float* Wi = (const float*)d_in[2];
  const float* bi = (const float*)d_in[3];
  const float* Wm = (const float*)d_in[4];
  const float* bm = (const float*)d_in[5];
  float* out = (float*)d_out;

  unsigned* cnt         = (unsigned*)d_ws;                          // barrier counter
  unsigned short* stage = (unsigned short*)((char*)d_ws + 256);     // 2 x 64 KB bf16

  hipMemsetAsync(d_ws, 0, 256, stream);  // reset barrier counter (graph-capturable)
  hipLaunchKernelGGL(fastslow_rnn_kernel, dim3(NWG), dim3(THREADS), 0, stream,
                     u, h0, Wi, bi, Wm, bm, out, stage, cnt);
}
